// DyGrEncoderModel_54443005444660
// MI455X (gfx1250) — compile-verified, matrix operand order checked by reading, not run on hardware
//
#include <hip/hip_runtime.h>
#include <hip/hip_bf16.h>

#define NNODES 50000
#define NEDGES 800000
#define FIN    64
#define CCH    100
#define TOUT   8

typedef __attribute__((ext_vector_type(16))) __bf16 v16bf;
typedef __attribute__((ext_vector_type(8)))  __bf16 v8bf;
typedef __attribute__((ext_vector_type(8)))  float  v8f;

__device__ __forceinline__ __bf16 f2bf(float f) {
  unsigned u = __builtin_bit_cast(unsigned, f);
  u += 0x7FFFu + ((u >> 16) & 1u);           // round-to-nearest-even to bf16
  unsigned short h = (unsigned short)(u >> 16);
  return __builtin_bit_cast(__bf16, h);
}

__device__ __forceinline__ float sigm(float x) { return 1.0f / (1.0f + __expf(-x)); }

// ---------------------------------------------------------------------------
// WMMA GEMM: D[M,Nn] = A[M,KP]bf16 @ B[Npad,KP]bf16^T (+bias, +=C).
// A,B are pre-padded bf16 (K padded to KP multiple of 32, B rows padded to
// tile multiple with zeros) -> branch-free inner loop: 4x b128 loads + wmma.
// One wave32 per 16x16 tile, 8 waves/block striped over M tiles.
// ---------------------------------------------------------------------------
template <int KP>
__global__ void wmma_gemm_bf16(const __bf16* __restrict__ A,
                               const __bf16* __restrict__ B,
                               float* __restrict__ Cm, int ldc,
                               int M, int Nn,
                               const float* __restrict__ bias,
                               int accumulate) {
  const int lane = threadIdx.x & 31;
  const int wave = threadIdx.x >> 5;
  const int mt = blockIdx.x * 8 + wave;
  const int nt = blockIdx.y;
  const int mtiles = (M + 15) >> 4;
  if (mt >= mtiles) return;                  // wave-uniform; EXEC stays full

  const int rsel  = lane & 15;               // A row / B col within tile
  const int khalf = (lane >> 4) << 3;        // lanes 16-31: K+8 / K+24 halves

  const __bf16* Ap = A + (long long)(mt * 16 + rsel) * KP + khalf;
  const __bf16* Bp = B + (long long)(nt * 16 + rsel) * KP + khalf;

  v8f acc = {};
#pragma unroll
  for (int kk = 0; kk < KP; kk += 32) {
    v8bf alo = *(const v8bf*)(Ap + kk);
    v8bf ahi = *(const v8bf*)(Ap + kk + 16);
    v8bf blo = *(const v8bf*)(Bp + kk);
    v8bf bhi = *(const v8bf*)(Bp + kk + 16);
    v16bf av = __builtin_shufflevector(alo, ahi, 0,1,2,3,4,5,6,7,8,9,10,11,12,13,14,15);
    v16bf bv = __builtin_shufflevector(blo, bhi, 0,1,2,3,4,5,6,7,8,9,10,11,12,13,14,15);
    acc = __builtin_amdgcn_wmma_f32_16x16x32_bf16(false, av, false, bv,
                                                  (short)0, acc, false, false);
  }

  // C/D layout: VGPR i -> M = i + 8*(lane>=16), N = lane&15
  const int col  = nt * 16 + (lane & 15);
  const int rofs = (lane >> 4) << 3;
  if (col < Nn) {
    float bval = bias ? bias[col] : 0.0f;
#pragma unroll
    for (int i = 0; i < 8; ++i) {
      int row = mt * 16 + rofs + i;
      if (row < M) {
        long long idx = (long long)row * ldc + col;
        float v = acc[i] + bval;
        if (accumulate) v += Cm[idx];
        Cm[idx] = v;
      }
    }
  }
}

__global__ void zero_f32(float* __restrict__ p, int n) {
  int i = blockIdx.x * blockDim.x + threadIdx.x;
  if (i < n) p[i] = 0.0f;
}

// f32 [R, ldsrc] -> bf16 [Rpad, KP], zero-padded rows/cols
__global__ void cvt_pad(const float* __restrict__ src, int ldsrc, int R, int K,
                        __bf16* __restrict__ dst, int Rpad, int KP) {
  int idx = blockIdx.x * blockDim.x + threadIdx.x;
  if (idx >= Rpad * KP) return;
  int r = idx / KP, k = idx % KP;
  float v = (r < R && k < K) ? src[(long long)r * ldsrc + k] : 0.0f;
  dst[idx] = f2bf(v);
}

// ggc_w[0][k,j] -> Wt_bf[j,k], j padded 100->112, k<64
__global__ void cvt_ggc_t(const float* __restrict__ W, __bf16* __restrict__ Wt) {
  int idx = blockIdx.x * blockDim.x + threadIdx.x;
  if (idx >= 112 * FIN) return;
  int j = idx / FIN, k = idx % FIN;
  float v = (j < CCH) ? W[k * CCH + j] : 0.0f;
  Wt[idx] = f2bf(v);
}

// 128 lanes per edge: coalesced gather of m[src,:], coalesced L2 atomics into summ[dst,:]
__global__ void scatter_mean_accum(const float* __restrict__ m,
                                   const int* __restrict__ ei,
                                   const float* __restrict__ ew,
                                   float* __restrict__ summ,
                                   float* __restrict__ cnt) {
  long long gid = (long long)blockIdx.x * blockDim.x + threadIdx.x;
  int e = (int)(gid >> 7);
  int c = (int)(gid & 127);
  if (e >= NEDGES) return;
  int s = ei[e];
  int d = ei[NEDGES + e];
  float w = ew[e];
  if (c < CCH) atomicAdd(&summ[(long long)d * CCH + c], m[(long long)s * CCH + c] * w);
  if (c == 0)  atomicAdd(&cnt[d], 1.0f);
}

// agg = summ / max(cnt,1), emitted directly as bf16 [N,128] (K padded)
__global__ void mean_cvt(const float* __restrict__ summ, const float* __restrict__ cnt,
                         __bf16* __restrict__ aggbf) {
  int idx = blockIdx.x * blockDim.x + threadIdx.x;
  if (idx >= NNODES * 128) return;
  int n = idx >> 7, j = idx & 127;
  float v = 0.0f;
  if (j < CCH) v = summ[(long long)n * CCH + j] / fmaxf(cnt[n], 1.0f);
  aggbf[idx] = f2bf(v);
}

// GRUCell pointwise -> conv_h emitted as bf16 [N,128] (K padded)
__global__ void gru_combine(const float* __restrict__ gi, const float* __restrict__ gh,
                            const float* __restrict__ x, __bf16* __restrict__ convhbf) {
  int idx = blockIdx.x * blockDim.x + threadIdx.x;
  if (idx >= NNODES * 128) return;
  int n = idx >> 7, j = idx & 127;
  float v = 0.0f;
  if (j < CCH) {
    const float* gin = gi + (long long)n * 3 * CCH;
    const float* ghn = gh + (long long)n * 3 * CCH;
    float r  = sigm(gin[j]            + ghn[j]);
    float z  = sigm(gin[CCH + j]      + ghn[CCH + j]);
    float nn = tanhf(gin[2 * CCH + j] + r * ghn[2 * CCH + j]);
    float xp = (j < FIN) ? x[(long long)n * FIN + j] : 0.0f;
    v = (1.0f - z) * nn + z * xp;
  }
  convhbf[idx] = f2bf(v);
}

// LSTM pointwise (i,f,g,o): h1/c1 f32 into d_out + relu(h1) as bf16 for the head
__global__ void lstm_combine(const float* __restrict__ gates, const float* __restrict__ c0,
                             float* __restrict__ h1, float* __restrict__ c1,
                             __bf16* __restrict__ h1bf) {
  int idx = blockIdx.x * blockDim.x + threadIdx.x;
  if (idx >= NNODES * FIN) return;
  int n = idx >> 6, j = idx & 63;
  const float* g = gates + (long long)n * 4 * FIN;
  float ig = sigm(g[j]);
  float fg = sigm(g[FIN + j]);
  float gg = tanhf(g[2 * FIN + j]);
  float og = sigm(g[3 * FIN + j]);
  float c = fg * c0[idx] + ig * gg;
  float h = og * tanhf(c);
  c1[idx] = c;
  h1[idx] = h;
  h1bf[idx] = f2bf(fmaxf(h, 0.0f));
}

extern "C" void kernel_launch(void* const* d_in, const int* in_sizes, int n_in,
                              void* d_out, int out_size, void* d_ws, size_t ws_size,
                              hipStream_t stream) {
  const float* x         = (const float*)d_in[0];
  const int*   ei        = (const int*)d_in[1];
  const float* ew        = (const float*)d_in[2];
  const float* h0        = (const float*)d_in[3];
  const float* c0        = (const float*)d_in[4];
  const float* ggc_w     = (const float*)d_in[5];
  const float* gru_w_ih  = (const float*)d_in[6];
  const float* gru_w_hh  = (const float*)d_in[7];
  const float* gru_b_ih  = (const float*)d_in[8];
  const float* gru_b_hh  = (const float*)d_in[9];
  const float* lstm_w_ih = (const float*)d_in[10];
  const float* lstm_w_hh = (const float*)d_in[11];
  const float* lstm_b_ih = (const float*)d_in[12];
  const float* lstm_b_hh = (const float*)d_in[13];
  const float* lin_w     = (const float*)d_in[14];
  const float* lin_b     = (const float*)d_in[15];

  // ---- carve 256B-aligned workspace chunks ----
  char* base = (char*)d_ws;
  auto alloc = [&](size_t bytes) -> char* {
    char* p = base;
    base += (bytes + 255) & ~(size_t)255;
    return p;
  };
  float*  m      = (float*)alloc((size_t)NNODES * CCH * 4);      // GGC projection
  float*  summ   = (float*)alloc((size_t)NNODES * CCH * 4);      // scatter accum
  float*  cnt    = (float*)alloc((size_t)NNODES * 4);            // degrees
  float*  gi     = (float*)alloc((size_t)NNODES * 3 * CCH * 4);  // reused as LSTM gates
  float*  gh     = (float*)alloc((size_t)NNODES * 3 * CCH * 4);
  __bf16* xbf    = (__bf16*)alloc((size_t)NNODES * 64 * 2);
  __bf16* h0bf   = (__bf16*)alloc((size_t)NNODES * 64 * 2);
  __bf16* aggbf  = (__bf16*)alloc((size_t)NNODES * 128 * 2);
  __bf16* chbf   = (__bf16*)alloc((size_t)NNODES * 128 * 2);
  __bf16* h1bf   = (__bf16*)alloc((size_t)NNODES * 64 * 2);
  __bf16* wtbf   = (__bf16*)alloc((size_t)112 * 64 * 2);         // ggc^T
  __bf16* gwihbf = (__bf16*)alloc((size_t)304 * 128 * 2);
  __bf16* gwhhbf = (__bf16*)alloc((size_t)304 * 64 * 2);
  __bf16* lwihbf = (__bf16*)alloc((size_t)256 * 128 * 2);
  __bf16* lwhhbf = (__bf16*)alloc((size_t)256 * 64 * 2);
  __bf16* linwbf = (__bf16*)alloc((size_t)16 * 64 * 2);

  float* out = (float*)d_out;                      // [N,8]
  float* h1  = out + (size_t)NNODES * TOUT;        // [N,64]
  float* c1  = h1  + (size_t)NNODES * FIN;         // [N,64]
  float* gates = gi;                               // reuse after GRU combine

  const int TPB = 256;
  auto blocks = [](long long n, int tpb) { return (unsigned)((n + tpb - 1) / tpb); };
  auto gemm_grid = [](int M, int Nn) {
    return dim3((unsigned)((((M + 15) / 16) + 7) / 8), (unsigned)((Nn + 15) / 16), 1);
  };

  // 0) zero scatter accumulators
  zero_f32<<<blocks((long long)NNODES * CCH, TPB), TPB, 0, stream>>>(summ, NNODES * CCH);
  zero_f32<<<blocks(NNODES, TPB), TPB, 0, stream>>>(cnt, NNODES);

  // 1) stage weights + activations as padded bf16
  cvt_ggc_t<<<blocks(112 * 64, TPB), TPB, 0, stream>>>(ggc_w, wtbf);
  cvt_pad<<<blocks(304 * 128, TPB), TPB, 0, stream>>>(gru_w_ih, CCH, 300, CCH, gwihbf, 304, 128);
  cvt_pad<<<blocks(304 * 64, TPB), TPB, 0, stream>>>(gru_w_hh, CCH, 300, 64, gwhhbf, 304, 64);
  cvt_pad<<<blocks(256 * 128, TPB), TPB, 0, stream>>>(lstm_w_ih, CCH, 256, CCH, lwihbf, 256, 128);
  cvt_pad<<<blocks(256 * 64, TPB), TPB, 0, stream>>>(lstm_w_hh, FIN, 256, FIN, lwhhbf, 256, 64);
  cvt_pad<<<blocks(16 * 64, TPB), TPB, 0, stream>>>(lin_w, FIN, TOUT, FIN, linwbf, 16, 64);
  cvt_pad<<<blocks((long long)NNODES * 64, TPB), TPB, 0, stream>>>(x, FIN, NNODES, FIN, xbf, NNODES, 64);
  cvt_pad<<<blocks((long long)NNODES * 64, TPB), TPB, 0, stream>>>(h0, FIN, NNODES, FIN, h0bf, NNODES, 64);

  // 2) m = xp @ ggc_w[0]  ==  x_bf[N,64] @ Wt_bf[112,64]^T
  wmma_gemm_bf16<64><<<gemm_grid(NNODES, CCH), TPB, 0, stream>>>(
      xbf, wtbf, m, CCH, NNODES, CCH, nullptr, 0);

  // 3) weighted scatter-add over edges + degree count (L2-resident atomics)
  scatter_mean_accum<<<blocks((long long)NEDGES * 128, TPB), TPB, 0, stream>>>(
      m, ei, ew, summ, cnt);

  // 4) agg = summ / max(cnt,1) -> bf16 [N,128]
  mean_cvt<<<blocks((long long)NNODES * 128, TPB), TPB, 0, stream>>>(summ, cnt, aggbf);

  // 5) gi = agg @ gru_w_ih^T + b_ih
  wmma_gemm_bf16<128><<<gemm_grid(NNODES, 3 * CCH), TPB, 0, stream>>>(
      aggbf, gwihbf, gi, 3 * CCH, NNODES, 3 * CCH, gru_b_ih, 0);

  // 6) gh = xp @ gru_w_hh^T + b_hh  (zero-padded cols collapse K to 64)
  wmma_gemm_bf16<64><<<gemm_grid(NNODES, 3 * CCH), TPB, 0, stream>>>(
      xbf, gwhhbf, gh, 3 * CCH, NNODES, 3 * CCH, gru_b_hh, 0);

  // 7) GRU pointwise -> conv_h (bf16, K padded to 128)
  gru_combine<<<blocks((long long)NNODES * 128, TPB), TPB, 0, stream>>>(gi, gh, x, chbf);

  // 8) gates = conv_h @ lstm_w_ih^T + b_ih
  wmma_gemm_bf16<128><<<gemm_grid(NNODES, 4 * FIN), TPB, 0, stream>>>(
      chbf, lwihbf, gates, 4 * FIN, NNODES, 4 * FIN, lstm_b_ih, 0);

  // 9) gates += h0 @ lstm_w_hh^T + b_hh
  wmma_gemm_bf16<64><<<gemm_grid(NNODES, 4 * FIN), TPB, 0, stream>>>(
      h0bf, lwhhbf, gates, 4 * FIN, NNODES, 4 * FIN, lstm_b_hh, 1);

  // 10) LSTM pointwise -> h1, c1 (d_out) + relu(h1) bf16
  lstm_combine<<<blocks((long long)NNODES * FIN, TPB), TPB, 0, stream>>>(
      gates, c0, h1, c1, h1bf);

  // 11) out = relu(h1) @ lin_w^T + lin_b
  wmma_gemm_bf16<64><<<gemm_grid(NNODES, TOUT), TPB, 0, stream>>>(
      h1bf, linwbf, out, TOUT, NNODES, TOUT, lin_b, 0);
}
